// TemporalInteraction_88845693485498
// MI455X (gfx1250) — matrix operand order repaired and checked by reading, not compiled, not run on hardware
//
#include <hip/hip_runtime.h>

typedef __attribute__((ext_vector_type(16))) _Float16 v16h;
typedef __attribute__((ext_vector_type(8)))  float    v8f;

#define BB 8
#define NN 2048
#define QQ 512

#define KMAXA 3072   // 16 rows x 192 halfs (largest layer input: combine1 K=192)
#define KMAXB 2048   // 16 rows x 128 halfs (largest hidden: 128)

union AFrag { v16h v; unsigned u[8]; };

// One WMMA GEMM layer: D[16 x NOUT] = relu?(Xin[16 x KPAD] * Wpacked + bias)
// Xin row-major f16 in LDS, Wpacked in pre-swizzled per-lane fragment order,
// output stored row-major f16 into xout (stride OUTS).
__device__ __forceinline__ void mlp_layer(
    const _Float16* xin, int KPAD,
    const _Float16* wpk, const float* bias, int NOUT,
    _Float16* xout, int OUTS, bool relu, int lane)
{
  const int hi  = lane >> 4;
  const int col = lane & 15;
  const int NT  = NOUT >> 4;
  const int KT  = KPAD >> 5;
  for (int nt = 0; nt < NT; ++nt) {
    v8f acc = {0.f,0.f,0.f,0.f,0.f,0.f,0.f,0.f};
    for (int kt = 0; kt < KT; ++kt) {
      AFrag a;
      const _Float16* xr = xin + col*KPAD + kt*32 + hi*8;
      #pragma unroll
      for (int j = 0; j < 8; ++j) {
        int kb = 2*j + ((j >= 4) ? 8 : 0);     // {0,2,4,6,16,18,20,22}
        a.u[j] = *(const unsigned*)(xr + kb);  // half-pair -> one 32b load
      }
      v16h bf = *(const v16h*)(wpk + (size_t)((kt*NT + nt)*32 + lane)*16);
      acc = __builtin_amdgcn_wmma_f32_16x16x32_f16(
                false, a.v, false, bf, (short)0, acc, false, false);
    }
    float bc = bias[nt*16 + col];
    #pragma unroll
    for (int vi = 0; vi < 8; ++vi) {
      float t = acc[vi] + bc;
      if (relu) t = fmaxf(t, 0.f);
      xout[(hi*8 + vi)*OUTS + nt*16 + col] = (_Float16)t;
    }
  }
}

// ---------------- weight packing: fp32 [KIN x NOUT] -> f16 B-fragments -----
__global__ void pack_w_kernel(const float* __restrict__ W, _Float16* __restrict__ out,
                              int KIN, int KPAD, int NOUT)
{
  int p = blockIdx.x*256 + threadIdx.x;
  int total = KPAD * NOUT;
  if (p >= total) return;
  int frag = p >> 9;          // 512 halfs per (kt,nt) fragment
  int rem  = p & 511;
  int lane = rem >> 4, e = rem & 15;
  int NT = NOUT >> 4;
  int nt = frag % NT, kt = frag / NT;
  int k = kt*32 + (lane >> 4)*8 + (e & 7) + (e >> 3)*16;     // B layout: lanes 0-15 K=0..7,16..23; 16-31 K=8..15,24..31
  int n = nt*16 + (lane & 15);
  float v = (k < KIN) ? W[(size_t)k*NOUT + n] : 0.f;
  out[p] = (_Float16)v;
}

// ---------------- sinusoidal time encoding --------------------------------
__global__ void time_encode_kernel(const float* __restrict__ t, float* __restrict__ out, int n)
{
  int i = blockIdx.x*256 + threadIdx.x;
  if (i >= n) return;
  float tv = t[i];
  const float w0 = 1.0f, w1 = 0.562341325f, w2 = 0.316227766f, w3 = 0.177827941f;
  float* o = out + (size_t)i*8;
  o[0] = __sinf(tv*w0); o[1] = __sinf(tv*w1); o[2] = __sinf(tv*w2); o[3] = __sinf(tv*w3);
  o[4] = __cosf(tv*w0); o[5] = __cosf(tv*w1); o[6] = __cosf(tv*w2); o[7] = __cosf(tv*w3);
}

// ---------------- top-k kernels (one thread per query) --------------------
__global__ void topk_space_kernel(const float* __restrict__ sp, int* __restrict__ out)
{
  int t = blockIdx.x*256 + threadIdx.x;
  if (t >= BB*NN) return;
  int b = t >> 11, q = t & (NN-1);
  const float* P = sp + (size_t)b*NN*3;
  float qx = P[q*3+0], qy = P[q*3+1], qz = P[q*3+2];
  float bd[16]; int bi[16];
  #pragma unroll
  for (int k = 0; k < 16; ++k) { bd[k] = 3.0e38f; bi[k] = 0; }
  for (int j = 0; j < NN; ++j) {
    float dx = P[j*3+0]-qx, dy = P[j*3+1]-qy, dz = P[j*3+2]-qz;
    float d = dx*dx + dy*dy + dz*dz;
    if (d < bd[15]) {
      int p = 15;
      while (p > 0 && bd[p-1] > d) { bd[p] = bd[p-1]; bi[p] = bi[p-1]; --p; }
      bd[p] = d; bi[p] = j;
    }
  }
  int* o = out + (size_t)t*16;
  #pragma unroll
  for (int k = 0; k < 16; ++k) o[k] = bi[k];
}

__global__ void topk_abs_kernel(const float* __restrict__ tq, const float* __restrict__ tk,
                                int NQ, int* __restrict__ out)
{
  int t = blockIdx.x*256 + threadIdx.x;
  if (t >= BB*NQ) return;
  int b = t / NQ, q = t - b*NQ;
  float qv = tq[(size_t)b*NQ + q];
  const float* K = tk + (size_t)b*NN;
  float bd[8]; int bi[8];
  #pragma unroll
  for (int k = 0; k < 8; ++k) { bd[k] = 3.0e38f; bi[k] = 0; }
  for (int j = 0; j < NN; ++j) {
    float d = fabsf(K[j] - qv);
    if (d < bd[7]) {
      int p = 7;
      while (p > 0 && bd[p-1] > d) { bd[p] = bd[p-1]; bi[p] = bi[p-1]; --p; }
      bd[p] = d; bi[p] = j;
    }
  }
  int* o = out + (size_t)t*8;
  #pragma unroll
  for (int k = 0; k < 8; ++k) o[k] = bi[k];
}

// ---------------- space neighborhood conv (K=16, 1 query per tile) --------
__global__ __launch_bounds__(128) void space_conv_kernel(
    const float* __restrict__ sp, const float* __restrict__ key, int C,
    const int* __restrict__ idx,
    const _Float16* wp0, const float* b0, int K0PAD,
    const _Float16* wp1, const float* b1,
    const _Float16* wp2, const float* b2,
    int H, int COUT, float* __restrict__ out)
{
  __shared__ _Float16 bufA[4][KMAXA];
  __shared__ _Float16 bufB[4][KMAXB];
  const int w = threadIdx.x >> 5, lane = threadIdx.x & 31;
  const int tile = blockIdx.x*4 + w;
  const int b = tile >> 11, q = tile & (NN-1);
  const size_t qb = (size_t)b*NN + q;
  const int* ix = idx + qb*16;
  const int KIN = 6 + C;
  _Float16* A  = bufA[w];
  _Float16* Bf = bufB[w];
  for (int i = lane; i < 16*K0PAD; i += 32) {
    int r = i / K0PAD, c = i - r*K0PAD;
    int nbr = ix[r];
    size_t nb = (size_t)b*NN + nbr;
    float v = 0.f;
    if (c < 3)        v = sp[qb*3 + c] - sp[nb*3 + c];   // rel_pos
    else if (c < 6)   v = sp[nb*3 + (c-3)];              // nbr space_pts (from space_in)
    else if (c < KIN) v = key[nb*C + (c-6)];             // nbr key feats
    A[i] = (_Float16)v;
  }
  __syncthreads();
  mlp_layer(A, K0PAD, wp0, b0, H, Bf, H, true, lane);
  __syncthreads();
  mlp_layer(Bf, H, wp1, b1, H, A, H, true, lane);
  __syncthreads();
  mlp_layer(A, H, wp2, b2, COUT, Bf, COUT, false, lane);
  __syncthreads();
  for (int c = lane; c < COUT; c += 32) {                // max over 16 neighbors
    float m = -3.0e38f;
    for (int r = 0; r < 16; ++r) m = fmaxf(m, (float)Bf[r*COUT + c]);
    out[qb*COUT + c] = m;
  }
}

// ---------------- time/target neighborhood conv (K=8, 2 queries per tile) -
__global__ __launch_bounds__(128) void time_conv_kernel(
    const float* __restrict__ enck, const float* __restrict__ encq, int NQ,
    const float* __restrict__ key, int C,
    const float* __restrict__ snei, int CS,
    const int* __restrict__ idx,
    const _Float16* wp0, const float* b0, int K0PAD,
    const _Float16* wp1, const float* b1,
    const _Float16* wp2, const float* b2,
    int H0, int H1, int COUT, float* __restrict__ out)
{
  __shared__ _Float16 bufA[4][KMAXA];
  __shared__ _Float16 bufB[4][KMAXB];
  const int w = threadIdx.x >> 5, lane = threadIdx.x & 31;
  const int tile = blockIdx.x*4 + w;
  const int halfq = NQ >> 1;
  const int b = tile / halfq;
  const int qbase = (tile - b*halfq)*2;
  const int KIN = 16 + C + CS;
  _Float16* A  = bufA[w];
  _Float16* Bf = bufB[w];
  for (int i = lane; i < 16*K0PAD; i += 32) {
    int r = i / K0PAD, c = i - r*K0PAD;
    int query = qbase + (r >> 3);
    size_t qi = (size_t)b*NQ + query;
    int nbr = idx[qi*8 + (r & 7)];
    size_t nb = (size_t)b*NN + nbr;
    float v = 0.f;
    if (c < 8)          v = encq[qi*8 + c] - enck[nb*8 + c]; // rel time enc
    else if (c < 16)    v = enck[nb*8 + (c-8)];              // nbr enc_t
    else if (c < 16+C)  v = key[nb*C + (c-16)];              // nbr key feats
    else if (c < KIN)   v = snei[nb*CS + (c-16-C)];          // nbr space_nei
    A[i] = (_Float16)v;
  }
  __syncthreads();
  mlp_layer(A, K0PAD, wp0, b0, H0, Bf, H0, true, lane);
  __syncthreads();
  mlp_layer(Bf, H0, wp1, b1, H1, A, H1, true, lane);
  __syncthreads();
  mlp_layer(A, H1, wp2, b2, COUT, Bf, COUT, false, lane);
  __syncthreads();
  for (int i = lane; i < 2*COUT; i += 32) {                  // max over 8 nbrs, 2 queries
    int qh = (i < COUT) ? 0 : 1;
    int c  = i - qh*COUT;
    float m = -3.0e38f;
    for (int r = qh*8; r < qh*8 + 8; ++r) m = fmaxf(m, (float)Bf[r*COUT + c]);
    out[((size_t)b*NQ + qbase + qh)*COUT + c] = m;
  }
}

// ---------------- pointwise combine MLP (16 nodes per tile) ---------------
__global__ __launch_bounds__(128) void combine_kernel(
    const float* __restrict__ key, int C,
    const float* __restrict__ snei, int CS,
    const float* __restrict__ tnei, int CT, int K0PAD,
    const _Float16* wp0, const float* b0, int H,
    const _Float16* wp1, const float* b1, int COUT,
    float* __restrict__ out)
{
  __shared__ _Float16 bufA[4][KMAXA];
  __shared__ _Float16 bufB[4][KMAXB];
  const int w = threadIdx.x >> 5, lane = threadIdx.x & 31;
  const int tile = blockIdx.x*4 + w;
  const int nodebase = tile*16;
  const int KIN = C + CS + CT;
  _Float16* A  = bufA[w];
  _Float16* Bf = bufB[w];
  for (int i = lane; i < 16*K0PAD; i += 32) {
    int r = i / K0PAD, c = i - r*K0PAD;
    size_t node = (size_t)(nodebase + r);
    float v = 0.f;
    if (c < C)          v = key[node*C + c];
    else if (c < C+CS)  v = snei[node*CS + (c-C)];
    else if (c < KIN)   v = tnei[node*CT + (c-C-CS)];
    A[i] = (_Float16)v;
  }
  __syncthreads();
  mlp_layer(A, K0PAD, wp0, b0, H, Bf, H, true, lane);
  __syncthreads();
  mlp_layer(Bf, H, wp1, b1, COUT, A, COUT, false, lane);
  __syncthreads();
  for (int i = lane; i < 16*COUT; i += 32) {
    int r = i / COUT, c = i - r*COUT;
    out[(size_t)(nodebase + r)*COUT + c] = (float)A[i];
  }
}

// --------------------------------------------------------------------------
extern "C" void kernel_launch(void* const* d_in, const int* in_sizes, int n_in,
                              void* d_out, int out_size, void* d_ws, size_t ws_size,
                              hipStream_t stream)
{
  (void)in_sizes; (void)n_in; (void)out_size; (void)ws_size;
  const float* data = (const float*)d_in[0];
  const float* sp   = (const float*)d_in[2];
  const float* tp   = (const float*)d_in[3];
  const float* qp   = (const float*)d_in[4];

  char* ws = (char*)d_ws;
  size_t off = 0;
  auto alloc = [&](size_t bytes) -> void* {
    void* p = ws + off;
    off += (bytes + 255) & ~(size_t)255;
    return p;
  };

  float* enc_t = (float*)alloc((size_t)BB*NN*8*4);
  float* enc_q = (float*)alloc((size_t)BB*QQ*8*4);
  int*   idx_s = (int*)alloc((size_t)BB*NN*16*4);
  int*   idx_t = (int*)alloc((size_t)BB*NN*8*4);
  int*   idx_q = (int*)alloc((size_t)BB*QQ*8*4);
  float* snei0 = (float*)alloc((size_t)BB*NN*32*4);
  float* tnei0 = (float*)alloc((size_t)BB*NN*32*4);
  float* key1  = (float*)alloc((size_t)BB*NN*64*4);
  float* snei1 = (float*)alloc((size_t)BB*NN*64*4);
  float* tnei1 = (float*)alloc((size_t)BB*NN*64*4);
  float* key2  = (float*)alloc((size_t)BB*NN*64*4);

  // {d_in index of W, KIN, NOUT}; bias is at index+1
  static const int wmeta[19][3] = {
    {5,22,64},{7,64,64},{9,64,32},        // space layer0
    {11,70,64},{13,64,64},{15,64,64},     // space layer1
    {17,64,64},{19,64,64},{21,64,32},     // time layer0
    {23,144,64},{25,64,64},{27,64,64},    // time layer1
    {29,80,128},{31,128,64},              // combine layer0
    {33,192,128},{35,128,64},             // combine layer1
    {37,80,128},{39,128,128},{41,128,64}  // target
  };
  _Float16* wp[19];
  for (int i = 0; i < 19; ++i) {
    int KIN = wmeta[i][1], NOUT = wmeta[i][2];
    int KPAD = (KIN + 31) & ~31;
    wp[i] = (_Float16*)alloc((size_t)KPAD*NOUT*2);
    int total = KPAD*NOUT;
    pack_w_kernel<<<(total + 255)/256, 256, 0, stream>>>(
        (const float*)d_in[wmeta[i][0]], wp[i], KIN, KPAD, NOUT);
  }
  auto bias = [&](int i) { return (const float*)d_in[wmeta[i][0] + 1]; };

  time_encode_kernel<<<(BB*NN + 255)/256, 256, 0, stream>>>(tp, enc_t, BB*NN);
  time_encode_kernel<<<(BB*QQ + 255)/256, 256, 0, stream>>>(qp, enc_q, BB*QQ);
  topk_space_kernel<<<(BB*NN + 255)/256, 256, 0, stream>>>(sp, idx_s);
  topk_abs_kernel<<<(BB*NN + 255)/256, 256, 0, stream>>>(tp, tp, NN, idx_t);
  topk_abs_kernel<<<(BB*QQ + 255)/256, 256, 0, stream>>>(qp, tp, QQ, idx_q);

  // layer 0
  space_conv_kernel<<<BB*NN/4, 128, 0, stream>>>(sp, data, 16, idx_s,
      wp[0], bias(0), 32, wp[1], bias(1), wp[2], bias(2), 64, 32, snei0);
  time_conv_kernel<<<BB*NN/2/4, 128, 0, stream>>>(enc_t, enc_t, NN, data, 16,
      snei0, 32, idx_t,
      wp[6], bias(6), 64, wp[7], bias(7), wp[8], bias(8), 64, 64, 32, tnei0);
  combine_kernel<<<BB*NN/16/4, 128, 0, stream>>>(data, 16, snei0, 32, tnei0, 32, 96,
      wp[12], bias(12), 128, wp[13], bias(13), 64, key1);

  // layer 1
  space_conv_kernel<<<BB*NN/4, 128, 0, stream>>>(sp, key1, 64, idx_s,
      wp[3], bias(3), 96, wp[4], bias(4), wp[5], bias(5), 64, 64, snei1);
  time_conv_kernel<<<BB*NN/2/4, 128, 0, stream>>>(enc_t, enc_t, NN, key1, 64,
      snei1, 64, idx_t,
      wp[9], bias(9), 160, wp[10], bias(10), wp[11], bias(11), 64, 64, 64, tnei1);
  combine_kernel<<<BB*NN/16/4, 128, 0, stream>>>(key1, 64, snei1, 64, tnei1, 64, 192,
      wp[14], bias(14), 128, wp[15], bias(15), 64, key2);

  // target conv -> d_out [B, Q, 64]
  time_conv_kernel<<<BB*QQ/2/4, 128, 0, stream>>>(enc_t, enc_q, QQ, key2, 64,
      enc_t /*unused, CS=0*/, 0, idx_q,
      wp[16], bias(16), 96, wp[17], bias(17), wp[18], bias(18), 128, 128, 64,
      (float*)d_out);
}
